// VectorQuantizer_50199577755663
// MI455X (gfx1250) — matrix operand order repaired and checked by reading, not compiled, not run on hardware
//
#include <hip/hip_runtime.h>
#include <hip/hip_bf16.h>
#include <math.h>

// ---------------------------------------------------------------------------
// VQ-VAE vector quantizer for gfx1250 (MI455X), wave32 + WMMA f16 (split hi/lo
// for fp32-faithful dot products) + async global->LDS double-buffered codebook
// staging (GLOBAL_LOAD_ASYNC_TO_LDS_B128 / ASYNCcnt).
//
// Workspace layout (requires >= ~2.01 MB):
//   [0, 1MB)        Ehi  : 1024x512 _Float16  (hi part of codebook)
//   [1MB, 2MB)      Elo  : 1024x512 _Float16  (lo residual)
//   [2MB, +4KB)     e_sq : 1024 float         (||e_k||^2)
//   [2MB+4KB,+4KB)  hist : 1024 int           (code usage histogram)
//   [2MB+8KB,+4B)   lacc : 1 float            (sum of row losses)
// ---------------------------------------------------------------------------

typedef _Float16 half16 __attribute__((ext_vector_type(16)));
typedef float    float8 __attribute__((ext_vector_type(8)));

#define NPIX   32768      // 32 * 32 * 32 pixels
#define KCODES 1024
#define DDIM   512
#define NUMEL  16777216   // NPIX * DDIM
#define CHUNK  32         // codes staged per LDS buffer

// --- prep: split codebook into f16 hi + lo residual ------------------------
__global__ void vq_prep_split(const float* __restrict__ E,
                              _Float16* __restrict__ Ehi,
                              _Float16* __restrict__ Elo, int n) {
  int i = blockIdx.x * 256 + threadIdx.x;
  if (i < n) {
    float x = E[i];
    _Float16 hx = (_Float16)x;
    Ehi[i] = hx;
    Elo[i] = (_Float16)(x - (float)hx);
  }
}

// --- prep: e_sq per code, zero hist + loss accumulator ---------------------
__global__ void vq_prep_esq(const float* __restrict__ E,
                            float* __restrict__ esq,
                            int* __restrict__ hist,
                            float* __restrict__ lacc) {
  int k = blockIdx.x * 256 + threadIdx.x;   // 0..1023
  const float* r = E + (size_t)k * DDIM;
  float s = 0.f;
  for (int d = 0; d < DDIM; ++d) s += r[d] * r[d];
  esq[k] = s;
  hist[k] = 0;
  if (k == 0) *lacc = 0.f;
}

// Issue this thread's share of one 64KB chunk (32 codes, hi+lo) as async
// global->LDS B128 transfers. GVS mode: 64-bit SGPR base + 32-bit VGPR offset.
__device__ __forceinline__ void issue_chunk_async(
    const _Float16* __restrict__ Ehi, const _Float16* __restrict__ Elo,
    int chunk, unsigned ldsH, unsigned ldsL, int tid)
{
  const char* gH = (const char*)(Ehi + (size_t)chunk * CHUNK * DDIM);
  const char* gL = (const char*)(Elo + (size_t)chunk * CHUNK * DDIM);
  #pragma unroll
  for (int i = 0; i < 8; ++i) {            // 8 * 256 threads * 16B = 32KB/array
    unsigned voff = (unsigned)(tid + i * 256) * 16u;
    unsigned dH = ldsH + voff;
    unsigned dL = ldsL + voff;
    asm volatile(
        "global_load_async_to_lds_b128 %0, %2, %3\n\t"
        "global_load_async_to_lds_b128 %1, %2, %4"
        :: "v"(dH), "v"(dL), "v"(voff), "s"(gH), "s"(gL)
        : "memory");
  }
}

// --- main: distances via WMMA, argmin, quantized writeback -----------------
__global__ __launch_bounds__(256)
void vq_main(const float* __restrict__ z,      // [32,512,32,32] NCHW
             const float* __restrict__ E,      // [1024,512]
             const _Float16* __restrict__ Ehi,
             const _Float16* __restrict__ Elo,
             const float* __restrict__ esqg,
             int* __restrict__ hist,
             float* __restrict__ lacc,
             float* __restrict__ out)          // z_q in NCHW
{
  // Double-buffered 32-code chunk of the (hi,lo) codebook:
  //   2 buffers x (hi 32KB + lo 32KB) = 128KB LDS (<= 320KB/WGP on CDNA5)
  __shared__ __align__(32) _Float16 sB[2][2][CHUNK * DDIM];

  const int tid  = threadIdx.x;
  const int lane = tid & 31;
  const int wave = tid >> 5;
  const int m    = lane & 15;   // row (A) / col (B) within 16-wide tile
  const int h    = lane >> 4;   // half-wave selector

  const int n0  = (blockIdx.x * 8 + wave) * 16;  // first pixel row of this wave
  const int b   = n0 >> 10;                      // batch index
  const int hw0 = n0 & 1023;                     // h*32+w base

  const unsigned ldsBuf0 = (unsigned)(size_t)(void*)&sB[0][0][0];
  const unsigned ldsBuf1 = (unsigned)(size_t)(void*)&sB[1][0][0];

  // Prologue: start DMA of chunk 0 into buffer 0; it overlaps the 256 global
  // loads that populate the resident A fragments below.
  issue_chunk_async(Ehi, Elo, 0, ldsBuf0, ldsBuf0 + CHUNK * DDIM * 2, tid);

  // ---- Load the full 16x512 A tile into registers as f16 hi/lo fragments.
  // 16-bit A 16x32 layout: lane (m,h); element e -> k = 16h+e (matches B)
  half16 Ahi[16], Alo[16];
  float ps = 0.f;   // partial sum of squares for row m (this lane's half of k)
  const float* zrow = z + ((size_t)b * DDIM) * 1024 + hw0 + m;
  #pragma unroll
  for (int dc = 0; dc < 16; ++dc) {
    #pragma unroll
    for (int e = 0; e < 16; ++e) {
      const int k = 16 * h + e;
      float x = zrow[(size_t)(dc * 32 + k) * 1024];
      ps += x * x;
      _Float16 xh = (_Float16)x;
      Ahi[dc][e] = xh;
      Alo[dc][e] = (_Float16)(x - (float)xh);
    }
  }
  // combine the two k-halves: lane m (and m+16) now holds full x_sq of row m
  float xsq = ps + __shfl_xor(ps, 16, 32);

  float minval[8];
  int   minidx[8];
  #pragma unroll
  for (int i = 0; i < 8; ++i) { minval[i] = 3.4e38f; minidx[i] = 0; }

  for (int chunk = 0; chunk < KCODES / CHUNK; ++chunk) {
    // Wait for this wave's DMA of chunk `chunk`, then barrier: after it, every
    // wave's async writes for this buffer are complete AND every wave has
    // finished reading the buffer that chunk+1's DMA will overwrite.
    asm volatile("s_wait_asynccnt 0x0" ::: "memory");
    __syncthreads();

    if (chunk + 1 < KCODES / CHUNK) {     // overlap next chunk's DMA w/ WMMAs
      unsigned ldsN = ((chunk + 1) & 1) ? ldsBuf1 : ldsBuf0;
      issue_chunk_async(Ehi, Elo, chunk + 1, ldsN, ldsN + CHUNK * DDIM * 2, tid);
    }

    const _Float16* bufH = &sB[chunk & 1][0][0];

    #pragma unroll
    for (int ct = 0; ct < 2; ++ct) {
      const int code0 = chunk * CHUNK + ct * 16;
      float8 acc = {};
      // 16-bit B 32x16 in this code's k order (same as A): lane (n=m,h); element e -> k = 16h+e
      // -> 16 contiguous f16 (32B) per fragment: clean ds_read_b128 pairs.
      #pragma unroll
      for (int dc = 0; dc < 16; ++dc) {
        const _Float16* pB = bufH + (ct * 16 + m) * DDIM + dc * 32 + 16 * h;
        half16 Bh = *(const half16*)pB;
        half16 Bl = *(const half16*)(pB + CHUNK * DDIM);  // lo plane is +32KB
        acc = __builtin_amdgcn_wmma_f32_16x16x32_f16(false, Ahi[dc], false, Bh,
                                                     (short)0, acc, false, false);
        acc = __builtin_amdgcn_wmma_f32_16x16x32_f16(false, Ahi[dc], false, Bl,
                                                     (short)0, acc, false, false);
        acc = __builtin_amdgcn_wmma_f32_16x16x32_f16(false, Alo[dc], false, Bh,
                                                     (short)0, acc, false, false);
      }
      // distance surrogate: d = e_sq - 2*dot  (x_sq added later for the loss)
      const int   code = code0 + m;         // column n of C tile
      const float esq  = esqg[code];
      #pragma unroll
      for (int i = 0; i < 8; ++i) {         // C layout: row = i + 8h, col = m
        float dv = esq - 2.0f * acc[i];
        if (dv < minval[i]) { minval[i] = dv; minidx[i] = code; }
      }
    }
  }

  // ---- reduce argmin across the 16 lanes of each half-wave
  #pragma unroll
  for (int i = 0; i < 8; ++i) {
    #pragma unroll
    for (int off = 8; off >= 1; off >>= 1) {
      float ov = __shfl_xor(minval[i], off, 32);
      int   oi = __shfl_xor(minidx[i], off, 32);
      if (ov < minval[i] || (ov == minval[i] && oi < minidx[i])) {
        minval[i] = ov; minidx[i] = oi;
      }
    }
  }

  // ---- gather per-row results to lane-indexed registers, emit stats
  int   myIdx = 0;
  float lloss = 0.f;
  #pragma unroll
  for (int mm = 0; mm < 16; ++mm) {
    const int src = (mm >> 3) << 4;                 // rows 0-7 from lanes 0-15
    const int i   = mm & 7;                         // rows 8-15 from lanes 16-31
    int   idx = __shfl(minidx[i], src, 32);
    float mv  = __shfl(minval[i], src, 32);
    float xs  = __shfl(xsq, mm, 32);
    if (m == mm)    myIdx = idx;                    // lanes mm and mm+16
    if (lane == mm) {                               // one lane per row
      lloss = xs + mv;                              // = ||x - e_idx||^2 (row)
      atomicAdd(&hist[idx], 1);
    }
  }
  #pragma unroll
  for (int off = 16; off >= 1; off >>= 1) lloss += __shfl_xor(lloss, off, 32);
  if (lane == 0) atomicAdd(lacc, lloss);

  // ---- write z_q back in NCHW; lanes 0-15 -> even d (64B contiguous stores),
  //      lanes 16-31 -> odd d. E rows gathered from L2-resident fp32 codebook.
  const float* Erow  = E + (size_t)myIdx * DDIM;
  float*       obase = out + ((size_t)b * DDIM) * 1024 + hw0 + m;
  for (int dd = h; dd < DDIM; dd += 2) {
    obase[(size_t)dd * 1024] = Erow[dd];
  }
}

// --- finalize: loss scalar + perplexity from histogram ---------------------
__global__ void vq_finalize(const int* __restrict__ hist,
                            const float* __restrict__ lacc,
                            float* __restrict__ out_tail) {
  __shared__ float red[256];
  float e = 0.f;
  for (int k = threadIdx.x; k < KCODES; k += 256) {
    float p = (float)hist[k] * (1.0f / (float)NPIX);
    e += p * logf(p + 1e-10f);
  }
  red[threadIdx.x] = e;
  __syncthreads();
  for (int s = 128; s >= 1; s >>= 1) {
    if (threadIdx.x < s) red[threadIdx.x] += red[threadIdx.x + s];
    __syncthreads();
  }
  if (threadIdx.x == 0) {
    out_tail[0] = lacc[0] * 1.25f / (float)NUMEL;  // z_q_loss + beta*z_e_loss
    out_tail[1] = expf(-red[0]);                   // perplexity
  }
}

extern "C" void kernel_launch(void* const* d_in, const int* in_sizes, int n_in,
                              void* d_out, int out_size, void* d_ws, size_t ws_size,
                              hipStream_t stream) {
  const float* z = (const float*)d_in[0];   // [32,512,32,32]
  const float* E = (const float*)d_in[1];   // [1024,512]

  char* ws = (char*)d_ws;
  _Float16* Ehi = (_Float16*)(ws);
  _Float16* Elo = (_Float16*)(ws + (1u << 20));
  float*    esq = (float*)(ws + (2u << 20));
  int*      hist = (int*)(ws + (2u << 20) + 4096);
  float*    lacc = (float*)(ws + (2u << 20) + 8192);
  float*    out = (float*)d_out;

  vq_prep_split<<<(KCODES * DDIM + 255) / 256, 256, 0, stream>>>(E, Ehi, Elo,
                                                                 KCODES * DDIM);
  vq_prep_esq<<<KCODES / 256, 256, 0, stream>>>(E, esq, hist, lacc);
  vq_main<<<NPIX / 128, 256, 0, stream>>>(z, E, Ehi, Elo, esq, hist, lacc, out);
  vq_finalize<<<1, 256, 0, stream>>>(hist, lacc, out + NUMEL);
}
